// NonlinearFilter_88278757802169
// MI455X (gfx1250) — matrix operand order repaired and checked by reading, not compiled, not run on hardware
//
#include <hip/hip_runtime.h>
#include <hip/hip_bf16.h>

// Problem dims (from reference): B=128, T=256, L=64, S=25, H=128
constexpr int NB = 128;
constexpr int NT = 256;
constexpr int NL = 64;
constexpr int NS = 25;
constexpr int NH = 128;

typedef _Float16 v16h __attribute__((ext_vector_type(16)));
typedef _Float16 v8h  __attribute__((ext_vector_type(8)));
typedef float    v8f  __attribute__((ext_vector_type(8)));

// LDS row strides (padded +8 halves = 16B so vector alignment holds and banks rotate)
constexpr int ZS = NL + 8;   // 72 halves
constexpr int HS = NH + 8;   // 136 halves

__device__ __forceinline__ v16h cat16(v8h lo, v8h hi) {
  return __builtin_shufflevector(lo, hi, 0,1,2,3,4,5,6,7,8,9,10,11,12,13,14,15);
}

// A fragment (16x32 f16), LDS row-major [row][stride].
// Lanes 0-15: row m0+lane, K = k0+{0..7, 16..23}; lanes 16-31: K = k0+{8..15, 24..31}.
__device__ __forceinline__ v16h loadA(const _Float16* base, int stride, int m0, int k0, int lane) {
  int row = m0 + (lane & 15);
  int kb  = k0 + ((lane >> 4) << 3);           // +0 or +8
  const _Float16* p = base + row * stride + kb;
  v8h lo = *(const v8h*)p;
  v8h hi = *(const v8h*)(p + 16);
  return cat16(lo, hi);
}

// B fragment (32x16 f16), weights stored transposed in LDS as [N][K].
// Lane n (0-15): col n0+n, K = k0+{0..7, 16..23}; lane n+16: same col, K = k0+{8..15, 24..31}.
__device__ __forceinline__ v16h loadB(const _Float16* base, int stride, int k0, int n0, int lane) {
  int col = n0 + (lane & 15);
  int kb  = k0 + ((lane >> 4) << 3);           // +0 or +8
  const _Float16* p = base + col * stride + kb;
  v8h lo = *(const v8h*)p;
  v8h hi = *(const v8h*)(p + 16);
  return cat16(lo, hi);
}

// ---- tanh: prefer CDNA5 hardware V_TANH_F32; else branch-free exp-based ----
#if defined(__has_builtin)
#  if __has_builtin(__builtin_amdgcn_tanhf)
#    define TANH_FAST(x) __builtin_amdgcn_tanhf(x)
#  elif __has_builtin(__builtin_amdgcn_tanh_f32)
#    define TANH_FAST(x) __builtin_amdgcn_tanh_f32(x)
#  endif
#endif
#ifndef TANH_FAST
__device__ __forceinline__ float tanh_fast_impl(float x) {
  // tanh(x) = 1 - 2/(exp(2x)+1); exp overflow->inf gives 1, underflow->0 gives -1.
  float e = __expf(2.0f * x);                  // single v_exp_f32, no branches
  return 1.0f - 2.0f / (e + 1.0f);
}
#  define TANH_FAST(x) tanh_fast_impl(x)
#endif

__device__ __forceinline__ float softplus_f(float x) {
  return (x > 20.f) ? x : logf(1.f + expf(x));
}

__global__ __launch_bounds__(128)
void nlf_kernel(const float* __restrict__ kin,   // (B,T,L)
                const float* __restrict__ Kin,   // (B,T,L)
                const float* __restrict__ nz,    // (T,S,B,L)
                const float* __restrict__ logQ,  // (L,)
                const float* __restrict__ m0v,   // (L,)
                const float* __restrict__ logQ0, // (L,)
                const float* __restrict__ W1,    // (L,H)
                const float* __restrict__ b1,    // (H,)
                const float* __restrict__ W2,    // (H,L)
                const float* __restrict__ b2,    // (L,)
                float* __restrict__ zf,          // (S,B,T,L)
                float* __restrict__ mf_o,        // (B,T,L)
                float* __restrict__ mp_o,
                float* __restrict__ Pf_o,
                float* __restrict__ Pp_o)
{
  __shared__ __align__(16) _Float16 W1t[NH][ZS];  // [h][l]  (B of matmul1, transposed)
  __shared__ __align__(16) _Float16 W2t[NL][HS];  // [l][h]  (B of matmul2, transposed)
  __shared__ __align__(16) _Float16 z16[32][ZS];  // ensemble, f16 A-matrix (rows 25..31 zero)
  __shared__ __align__(16) _Float16 h16[32][HS];  // tanh hidden, f16 A-matrix
  __shared__ float z32[32][NL + 1];               // ensemble, f32 (skip connection)
  __shared__ float b1s[NH], b2s[NL], Qd[NL], mfs[NL], sqs[NL];

  const int tid  = threadIdx.x;
  const int lane = tid & 31;
  const int w    = tid >> 5;     // wave id 0..3
  const int b    = blockIdx.x;

  // ---- one-time setup: weights -> f16 transposed LDS, biases, Q ----
  for (int i = tid; i < NL * NH; i += 128) { int l = i >> 7, h = i & (NH - 1); W1t[h][l] = (_Float16)W1[i]; }
  for (int i = tid; i < NH * NL; i += 128) { int h = i >> 6, l = i & (NL - 1); W2t[l][h] = (_Float16)W2[i]; }
  if (tid < NH) b1s[tid] = b1[tid];
  if (tid < NL) { b2s[tid] = b2[tid]; Qd[tid] = softplus_f(logQ[tid]); }
  for (int i = tid; i < 7 * ZS; i += 128) z16[NS + i / ZS][i % ZS] = (_Float16)0.f;

  // ---- t = 0 filter step ----
  if (tid < NL) {
    int l = tid;
    float P0 = softplus_f(logQ0[l]);
    float J0 = 1.f / P0;
    float h0 = J0 * m0v[l];
    size_t gi = (size_t)b * NT * NL + l;         // t = 0
    float Jf = J0 + Kin[gi];
    float Pf = 1.f / Jf;
    float mf = Pf * (h0 + kin[gi]);
    mf_o[gi] = mf; mp_o[gi] = m0v[l]; Pf_o[gi] = Pf; Pp_o[gi] = P0;
    mfs[l] = mf; sqs[l] = sqrtf(Pf);
  }
  __syncthreads();

  for (int e = tid; e < NS * NL; e += 128) {
    int s = e >> 6, l = e & (NL - 1);
    float wn = nz[((size_t)s * NB + b) * NL + l];               // t = 0
    float zv = mfs[l] + sqs[l] * wn;
    z32[s][l] = zv; z16[s][l] = (_Float16)zv;
    zf[(((size_t)s * NB + b) * NT) * NL + l] = zv;              // t = 0
  }
  __syncthreads();

  // ---- loop-invariant B fragments into VGPRs ----
  v16h B1f[2][2];                               // [local n-tile][k-chunk], N-tiles {2w, 2w+1}
  #pragma unroll
  for (int nt = 0; nt < 2; ++nt)
    #pragma unroll
    for (int kc = 0; kc < 2; ++kc)
      B1f[nt][kc] = loadB(&W1t[0][0], ZS, kc * 32, (2 * w + nt) * 16, lane);
  v16h B2f[4];                                  // N-tile = w
  #pragma unroll
  for (int kc = 0; kc < 4; ++kc)
    B2f[kc] = loadB(&W2t[0][0], HS, kc * 32, w * 16, lane);

  const v8f zero8 = {0.f, 0.f, 0.f, 0.f, 0.f, 0.f, 0.f, 0.f};
  const int hi8  = (lane >> 4) << 3;            // row offset 0 or 8 in C/D layout
  const int colH = lane & 15;                   // within-tile column
  const int lcol = w * 16 + colH;               // this lane's L-column

  // ---- sequential scan over time ----
  for (int t = 1; t < NT; ++t) {
    // prefetch next step's noise slab while we compute (global_prefetch_b8)
    if (t + 1 < NT && tid < NS)
      __builtin_prefetch(&nz[(((size_t)(t + 1) * NS + tid) * NB + b) * NL], 0, 1);

    // matmul1: z(32x64) @ W1(64x128); per wave N-tiles {2w,2w+1}
    v8f acc[2][2] = {{zero8, zero8}, {zero8, zero8}};
    #pragma unroll
    for (int kc = 0; kc < 2; ++kc) {
      v16h a0 = loadA(&z16[0][0], ZS, 0,  kc * 32, lane);
      v16h a1 = loadA(&z16[0][0], ZS, 16, kc * 32, lane);
      #pragma unroll
      for (int nt = 0; nt < 2; ++nt) {
        acc[0][nt] = __builtin_amdgcn_wmma_f32_16x16x32_f16(false, a0, false, B1f[nt][kc], (short)0, acc[0][nt], false, false);
        acc[1][nt] = __builtin_amdgcn_wmma_f32_16x16x32_f16(false, a1, false, B1f[nt][kc], (short)0, acc[1][nt], false, false);
      }
    }
    // bias + tanh -> h16  (TRANS v_tanh co-executes with WMMA issue)
    #pragma unroll
    for (int mt = 0; mt < 2; ++mt)
      #pragma unroll
      for (int nt = 0; nt < 2; ++nt) {
        int col = (2 * w + nt) * 16 + colH;
        float bb = b1s[col];
        #pragma unroll
        for (int j = 0; j < 8; ++j) {
          int r = mt * 16 + hi8 + j;
          h16[r][col] = (_Float16)TANH_FAST(acc[mt][nt][j] + bb);
        }
      }
    __syncthreads();

    // matmul2: h(32x128) @ W2(128x64); per wave N-tile = w
    v8f acc2[2] = {zero8, zero8};
    #pragma unroll
    for (int kc = 0; kc < 4; ++kc) {
      v16h a0 = loadA(&h16[0][0], HS, 0,  kc * 32, lane);
      v16h a1 = loadA(&h16[0][0], HS, 16, kc * 32, lane);
      acc2[0] = __builtin_amdgcn_wmma_f32_16x16x32_f16(false, a0, false, B2f[kc], (short)0, acc2[0], false, false);
      acc2[1] = __builtin_amdgcn_wmma_f32_16x16x32_f16(false, a1, false, B2f[kc], (short)0, acc2[1], false, false);
    }

    // m_th = z + mlp(z) + b2 ; ensemble moments over S (each wave owns whole columns)
    float bb2 = b2s[lcol];
    float s1 = 0.f, s2 = 0.f;
    #pragma unroll
    for (int mt = 0; mt < 2; ++mt)
      #pragma unroll
      for (int j = 0; j < 8; ++j) {
        int r = mt * 16 + hi8 + j;
        if (r < NS) {
          float v = z32[r][lcol] + acc2[mt][j] + bb2;
          s1 += v; s2 += v * v;
        }
      }
    s1 += __shfl_xor(s1, 16);
    s2 += __shfl_xor(s2, 16);

    if (lane < 16) {
      size_t gi = ((size_t)b * NT + t) * NL + lcol;
      float kt = kin[gi], Kt = Kin[gi];
      float mp = s1 * (1.f / NS);
      float Pp = Qd[lcol] + s2 * (1.f / NS) - mp * mp;
      float Jp = 1.f / Pp;
      float Jf = Jp + Kt;
      float Pf = 1.f / Jf;
      float mf = Pf * (Jp * mp + kt);
      mf_o[gi] = mf; mp_o[gi] = mp; Pf_o[gi] = Pf; Pp_o[gi] = Pp;
      mfs[lcol] = mf; sqs[lcol] = sqrtf(Pf);
    }
    __syncthreads();

    // sample next ensemble, stream z_f out
    for (int e = tid; e < NS * NL; e += 128) {
      int s = e >> 6, l2 = e & (NL - 1);
      float wn = nz[(((size_t)t * NS + s) * NB + b) * NL + l2];
      float zv = mfs[l2] + sqs[l2] * wn;
      z32[s][l2] = zv; z16[s][l2] = (_Float16)zv;
      zf[(((size_t)s * NB + b) * NT + t) * NL + l2] = zv;
    }
    __syncthreads();
  }
}

extern "C" void kernel_launch(void* const* d_in, const int* in_sizes, int n_in,
                              void* d_out, int out_size, void* d_ws, size_t ws_size,
                              hipStream_t stream) {
  (void)in_sizes; (void)n_in; (void)out_size; (void)d_ws; (void)ws_size;
  const float* kin   = (const float*)d_in[0];
  const float* Kin   = (const float*)d_in[1];
  const float* nz    = (const float*)d_in[2];
  const float* logQ  = (const float*)d_in[3];
  const float* m0v   = (const float*)d_in[4];
  const float* logQ0 = (const float*)d_in[5];
  const float* W1    = (const float*)d_in[6];
  const float* b1    = (const float*)d_in[7];
  const float* W2    = (const float*)d_in[8];
  const float* b2    = (const float*)d_in[9];

  float* out = (float*)d_out;
  const size_t nzf = (size_t)NS * NB * NT * NL;   // z_f
  const size_t nbt = (size_t)NB * NT * NL;        // per-(B,T,L) outputs
  float* zf  = out;
  float* mf  = out + nzf;
  float* mp  = mf + nbt;
  float* Pf  = mp + nbt;
  float* Pp  = Pf + nbt;

  nlf_kernel<<<dim3(NB), dim3(128), 0, stream>>>(kin, Kin, nz, logQ, m0v, logQ0,
                                                 W1, b1, W2, b2,
                                                 zf, mf, mp, Pf, Pp);
}
